// Attention_38697655337033
// MI455X (gfx1250) — matrix operand order repaired and checked by reading, not compiled, not run on hardware
//
#include <hip/hip_runtime.h>
#include <hip/hip_bf16.h>
#include <cmath>
#include <cstdint>

// Problem constants (from reference)
#define S_   1024
#define E_   2880
#define H_   64
#define KV_  8
#define D_   64
#define WIN_ 128
#define QK_DIM 4096   // H*D
#define KV_DIM 512    // KV*D
#define SCALING 0.125f
#define MAXC 160      // max window cols (9 blocks * 16), padded to mult of 32

typedef __attribute__((ext_vector_type(16))) __bf16 v16bf;
typedef __attribute__((ext_vector_type(8)))  float  v8f;
typedef uint32_t u32x4 __attribute__((ext_vector_type(4)));

union Frag16 {
  u32x4    q[2];
  uint32_t u[8];
  uint16_t h[16];
  v16bf    v;
};

__device__ __forceinline__ uint16_t f2bf(float f) {
  uint32_t u = __float_as_uint(f);
  u += 0x7FFFu + ((u >> 16) & 1u);   // round-to-nearest-even
  return (uint16_t)(u >> 16);
}

// ---------------------------------------------------------------- fp32->bf16
__global__ void cvt_bf16(const float* __restrict__ src,
                         uint16_t* __restrict__ dst, int n) {
  int i = blockIdx.x * blockDim.x + threadIdx.x;
  if (i < n) dst[i] = f2bf(src[i]);
}

// ------------------------------------------------------- GEMM: C = A*W^T + b
// A: [M,K] bf16 (row major), W: [N,K] bf16 (row major) -> NT GEMM, both
// operands K-contiguous, matching WMMA A/B fragment layouts directly.
// Block: 256 threads = 8 waves; block tile 16M x 512N; wave tile 16M x 64N.
// A tile is staged into a double-buffered LDS tile with the CDNA5
// async-to-LDS DMA path (global_load_async_to_lds_b32, ASYNCcnt).
// N must be a multiple of 64: out-of-range waves clamp to the last valid
// N-strip (redundant compute, no divergence) and skip their stores.
__global__ __launch_bounds__(256) void gemm_bias_nt(
    const uint16_t* __restrict__ A,
    const uint16_t* __restrict__ W,
    const float*    __restrict__ bias,
    float*          __restrict__ C,
    int M, int N, int K)
{
  __shared__ uint16_t lds_a[2][16 * 32];

  const int m0   = blockIdx.x * 16;
  const int t    = threadIdx.x;
  const int wave = t >> 5;
  const int lane = t & 31;
  const int nRaw = blockIdx.y * 512 + wave * 64;
  const bool active = (nRaw < N);
  const int n0   = active ? nRaw : (N - 64);   // clamp: keep all waves converged
  const int halfRow = lane & 15;
  const int hi      = lane >> 4;

  // staging coordinates: 256 threads x one b32 (2 bf16) = full 16x32 tile
  const int srow = t >> 4;          // 0..15
  const int scol = (t & 15) << 1;   // 0..30 even

  v8f acc[4] = {};

  // async copy of one A tile (16x32 bf16) into LDS buffer `buf`
  auto stage = [&](int buf, int k0) {
    uint32_t laddr = (uint32_t)(uintptr_t)&lds_a[buf][srow * 32 + scol];
    const uint16_t* g = A + (size_t)(m0 + srow) * K + k0 + scol;
    asm volatile("global_load_async_to_lds_b32 %0, %1, off"
                 :: "v"(laddr), "v"(g) : "memory");
  };

  stage(0, 0);
  asm volatile("s_wait_asynccnt 0x0" ::: "memory");
  __syncthreads();

  int buf = 0;
  for (int k0 = 0; k0 < K; k0 += 32) {
    // prefetch next A tile into the other buffer while we compute
    if (k0 + 32 < K) stage(buf ^ 1, k0 + 32);

    // A fragment: lane<16 -> M=halfRow, K {0..7,16..23}; lane>=16 -> {8..15,24..31}
    Frag16 af;
    {
      int kb = hi ? 8 : 0;
      af.q[0] = *(const u32x4*)&lds_a[buf][halfRow * 32 + kb];
      af.q[1] = *(const u32x4*)&lds_a[buf][halfRow * 32 + kb + 16];
    }
    // load all four B fragments first, then issue the WMMAs back-to-back
    Frag16 bfr[4];
    for (int j = 0; j < 4; ++j) {
      int col = n0 + j * 16 + halfRow;           // B column (N index)
      int kb  = k0 + (hi ? 8 : 0);               // lane half covers K {0..7,16..23} / {8..15,24..31}
      const u32x4* p = (const u32x4*)(W + (size_t)col * K + kb);
      bfr[j].q[0] = p[0];
      bfr[j].q[1] = p[2];
      __builtin_prefetch(W + (size_t)col * K + kb + 128, 0, 3);
    }
    for (int j = 0; j < 4; ++j)
      acc[j] = __builtin_amdgcn_wmma_f32_16x16x32_bf16(
          false, af.v, false, bfr[j].v, (short)0, acc[j], false, false);

    // own async copies done + all waves finished copying/reading
    asm volatile("s_wait_asynccnt 0x0" ::: "memory");
    __syncthreads();
    buf ^= 1;
  }

  if (active) {
    for (int j = 0; j < 4; ++j) {
      int n = n0 + j * 16 + halfRow;
      float b = bias[n];
      for (int r = 0; r < 8; ++r) {
        int m = m0 + r + hi * 8;
        C[(size_t)m * N + n] = acc[j][r] + b;
      }
    }
  }
}

// ------------------------------------------------- YaRN RoPE + bf16 convert
// grid (S, H+KV), block 32; lane = rotary pair index i in [0,32)
__global__ __launch_bounds__(32) void rope_cvt(
    const float* __restrict__ Qf, const float* __restrict__ Kf,
    const int*   __restrict__ positions,
    uint16_t* __restrict__ Qb, uint16_t* __restrict__ Kb,
    float low, float high, float mscale)
{
  int s  = blockIdx.x;
  int hh = blockIdx.y;
  int i  = threadIdx.x;

  float pf   = __powf(150000.0f, (2.0f * i) / 64.0f);
  float ramp = fminf(fmaxf(((float)i - low) / (high - low), 0.0f), 1.0f);
  float inv  = ramp / (32.0f * pf) + (1.0f - ramp) / pf;
  float ang  = (float)positions[s] * inv;
  float c  = __cosf(ang) * mscale;
  float sn = __sinf(ang) * mscale;

  const float* src;
  uint16_t* dst;
  if (hh < H_) {
    src = Qf + ((size_t)s * H_ + hh) * D_;
    dst = Qb + ((size_t)s * H_ + hh) * D_;
  } else {
    int kvh = hh - H_;
    src = Kf + ((size_t)s * KV_ + kvh) * D_;
    dst = Kb + ((size_t)s * KV_ + kvh) * D_;
  }
  float x1 = src[i], x2 = src[i + 32];
  dst[i]      = f2bf(x1 * c - x2 * sn);
  dst[i + 32] = f2bf(x2 * c + x1 * sn);
}

// --------------------------------------- windowed attention with sink (WMMA)
// One wave per (16-query block, head). grid (S/16, H), block 32.
__global__ __launch_bounds__(32) void attn_win(
    const uint16_t* __restrict__ Qb,   // [S,H,D]
    const uint16_t* __restrict__ Kb,   // [S,KV,D]
    const uint16_t* __restrict__ Vb,   // [S,KV,D]
    const float*    __restrict__ sinks,// [H]
    float*          __restrict__ O)    // [S,H*D]
{
  __shared__ float    sc[16][MAXC];
  __shared__ uint16_t pb[16][MAXC];

  const int qb = blockIdx.x, h = blockIdx.y;
  const int kvh = h >> 3;                 // H/KV = 8
  const int lane = threadIdx.x;
  const int row = lane & 15, hi = lane >> 4;
  const int q0  = qb * 16;
  const int kb0 = max(0, q0 - (WIN_ - 1)) >> 4;
  const int nb  = qb - kb0 + 1;           // <= 9 key blocks
  const int nk  = nb * 16;                // <= 144 keys

  // Q fragments (K dim = d): aq0 covers d 0..31, aq1 covers d 32..63
  Frag16 aq0, aq1;
  {
    const uint16_t* qrow = Qb + (((size_t)(q0 + row)) * H_ + h) * D_;
    int kb = hi ? 8 : 0;
    aq0.q[0] = *(const u32x4*)(qrow + kb);
    aq0.q[1] = *(const u32x4*)(qrow + kb + 16);
    aq1.q[0] = *(const u32x4*)(qrow + 32 + kb);
    aq1.q[1] = *(const u32x4*)(qrow + 32 + kb + 16);
  }

  // scores = (Q . K^T) * SCALING, masked, into LDS
  for (int b = 0; b < nb; ++b) {
    int kg = (kb0 + b) * 16;
    const uint16_t* krow = Kb + (((size_t)(kg + row)) * KV_ + kvh) * D_;
    int db = hi ? 8 : 0;
    Frag16 bk0, bk1;
    bk0.q[0] = *(const u32x4*)(krow + db);
    bk0.q[1] = *(const u32x4*)(krow + db + 16);
    bk1.q[0] = *(const u32x4*)(krow + 32 + db);
    bk1.q[1] = *(const u32x4*)(krow + 32 + db + 16);
    v8f c = {};
    c = __builtin_amdgcn_wmma_f32_16x16x32_bf16(false, aq0.v, false, bk0.v,
                                                (short)0, c, false, false);
    c = __builtin_amdgcn_wmma_f32_16x16x32_bf16(false, aq1.v, false, bk1.v,
                                                (short)0, c, false, false);
    for (int r = 0; r < 8; ++r) {
      int m = r + hi * 8;                 // local query row
      int q = q0 + m;
      int key = kg + row;                 // N index = lane&15
      float s = c[r] * SCALING;
      bool ok = (key <= q) && (q - key < WIN_);
      sc[m][b * 16 + row] = ok ? s : -1e30f;
    }
  }
  __syncthreads();

  // softmax with sink column (dropped after normalization): lanes 0..15 own rows
  if (lane < 16) {
    float snk = sinks[h];
    float m = snk;
    for (int c2 = 0; c2 < nk; ++c2) m = fmaxf(m, sc[lane][c2]);
    float denom = __expf(snk - m);
    for (int c2 = 0; c2 < nk; ++c2) {
      float e = __expf(sc[lane][c2] - m);
      sc[lane][c2] = e;
      denom += e;
    }
    float inv = 1.0f / denom;
    for (int c2 = 0; c2 < MAXC; ++c2)
      pb[lane][c2] = f2bf((c2 < nk) ? sc[lane][c2] * inv : 0.0f);
  }
  __syncthreads();

  // O = P @ V   (P: 16 x nk bf16 in LDS, V: keys x 64)
  v8f o[4] = {};
  const int nkc = (nk + 31) >> 5;
  for (int kc = 0; kc < nkc; ++kc) {
    Frag16 ap;
    int kb = kc * 32 + (hi ? 8 : 0);
    ap.q[0] = *(const u32x4*)&pb[row][kb];
    ap.q[1] = *(const u32x4*)&pb[row][kb + 16];
    int keybase = kb0 * 16 + kc * 32 + (hi ? 8 : 0);
    for (int j = 0; j < 4; ++j) {
      Frag16 bv;
      int d = j * 16 + row;
      for (int e = 0; e < 16; ++e) {
        int key = keybase + e + (e >= 8 ? 8 : 0);
        if (key > S_ - 1) key = S_ - 1;   // pad cols have P==0
        bv.h[e] = Vb[((size_t)key * KV_ + kvh) * D_ + d];
      }
      o[j] = __builtin_amdgcn_wmma_f32_16x16x32_bf16(
          false, ap.v, false, bv.v, (short)0, o[j], false, false);
    }
  }

  for (int j = 0; j < 4; ++j)
    for (int r = 0; r < 8; ++r) {
      int m = r + hi * 8;
      int d = j * 16 + row;
      O[((size_t)(q0 + m)) * QK_DIM + h * D_ + d] = o[j][r];
    }
}

// --------------------------------------------------------------------- host
extern "C" void kernel_launch(void* const* d_in, const int* in_sizes, int n_in,
                              void* d_out, int out_size, void* d_ws, size_t ws_size,
                              hipStream_t stream) {
  (void)in_sizes; (void)n_in; (void)out_size; (void)ws_size;
  const float* x    = (const float*)d_in[0];
  const int*   pos  = (const int*)  d_in[1];
  const float* Wq   = (const float*)d_in[2];
  const float* bq   = (const float*)d_in[3];
  const float* Wk   = (const float*)d_in[4];
  const float* bk   = (const float*)d_in[5];
  const float* Wv   = (const float*)d_in[6];
  const float* bv   = (const float*)d_in[7];
  const float* Wo   = (const float*)d_in[8];
  const float* bo   = (const float*)d_in[9];
  const float* sinks= (const float*)d_in[10];
  float* out = (float*)d_out;

  char* ws = (char*)d_ws;
  size_t off = 0;
  auto alloc = [&](size_t bytes) -> void* {
    void* p = ws + off;
    off = (off + bytes + 255) & ~(size_t)255;
    return p;
  };

  uint16_t* Xbf  = (uint16_t*)alloc((size_t)S_ * E_ * 2);
  uint16_t* Wqbf = (uint16_t*)alloc((size_t)QK_DIM * E_ * 2);
  uint16_t* Wkbf = (uint16_t*)alloc((size_t)KV_DIM * E_ * 2);
  uint16_t* Wvbf = (uint16_t*)alloc((size_t)KV_DIM * E_ * 2);
  uint16_t* Wobf = (uint16_t*)alloc((size_t)E_ * QK_DIM * 2);
  float*    Qf   = (float*)   alloc((size_t)S_ * QK_DIM * 4);
  float*    Kf   = (float*)   alloc((size_t)S_ * KV_DIM * 4);
  float*    Vf   = (float*)   alloc((size_t)S_ * KV_DIM * 4);
  uint16_t* Qbf  = (uint16_t*)alloc((size_t)S_ * QK_DIM * 2);
  uint16_t* Kbf  = (uint16_t*)alloc((size_t)S_ * KV_DIM * 2);
  uint16_t* Vbf  = (uint16_t*)alloc((size_t)S_ * KV_DIM * 2);
  float*    Of   = (float*)   alloc((size_t)S_ * QK_DIM * 4);
  uint16_t* Obf  = (uint16_t*)alloc((size_t)S_ * QK_DIM * 2);

  auto cvt = [&](const float* s, uint16_t* d, int n) {
    cvt_bf16<<<(n + 255) / 256, 256, 0, stream>>>(s, d, n);
  };

  cvt(x,  Xbf,  S_ * E_);
  cvt(Wq, Wqbf, QK_DIM * E_);
  cvt(Wk, Wkbf, KV_DIM * E_);
  cvt(Wv, Wvbf, KV_DIM * E_);
  cvt(Wo, Wobf, E_ * QK_DIM);

  // projections
  gemm_bias_nt<<<dim3(S_ / 16, QK_DIM / 512), 256, 0, stream>>>(
      Xbf, Wqbf, bq, Qf, S_, QK_DIM, E_);
  gemm_bias_nt<<<dim3(S_ / 16, 1), 256, 0, stream>>>(
      Xbf, Wkbf, bk, Kf, S_, KV_DIM, E_);
  gemm_bias_nt<<<dim3(S_ / 16, 1), 256, 0, stream>>>(
      Xbf, Wvbf, bv, Vf, S_, KV_DIM, E_);

  // YaRN ramp constants (host precompute, mirrors reference _rope_tables)
  const double PI = 3.14159265358979323846;
  auto find_dim = [&](double rot) {
    return 64.0 * std::log(4096.0 / (rot * 2.0 * PI)) / (2.0 * std::log(150000.0));
  };
  float low  = (float)std::max(find_dim(32.0), 0.0);
  float high = (float)std::min(find_dim(1.0), 31.0);
  if (low == high) high += 0.001f;
  float mscale = (float)(0.1 * std::log(32.0) + 1.0);

  rope_cvt<<<dim3(S_, H_ + KV_), 32, 0, stream>>>(Qf, Kf, pos, Qbf, Kbf,
                                                  low, high, mscale);
  cvt(Vf, Vbf, S_ * KV_DIM);

  attn_win<<<dim3(S_ / 16, H_), 32, 0, stream>>>(Qbf, Kbf, Vbf, sinks, Of);

  cvt(Of, Obf, S_ * QK_DIM);
  gemm_bias_nt<<<dim3(S_ / 16, (E_ + 511) / 512), 256, 0, stream>>>(
      Obf, Wobf, bo, out, S_, E_, QK_DIM);
}
